// CIKAConv_1340029796551
// MI455X (gfx1250) — compile-verified, matrix operand order checked by reading, not run on hardware
//
#include <hip/hip_runtime.h>
#include <hip/hip_bf16.h>

typedef __attribute__((ext_vector_type(16))) _Float16 v16h;
typedef __attribute__((ext_vector_type(8)))  _Float16 v8h;
typedef __attribute__((ext_vector_type(8)))  float    v8f;
typedef __attribute__((ext_vector_type(4)))  float    v4f;

#define CHN   32
#define IMG   256
#define TH    4
#define TW    16
#define HALO  2
#define TROWS (TH + 2*HALO)      // 8
#define TCOLS (TW + 2*HALO)      // 20
#define TELEM (CHN*TROWS*TCOLS)  // 5120

// Padded f16 B-matrix staging offsets (halves)
#define OB_K1 0        // [16][32]
#define OB_K2 512      // [32][32]
#define OB_S1 1536     // [64][32]
#define OB_S2 3584     // [32][64]
#define OB_LO 5632     // [32][32]
#define OB_UP 6656     // [32][32]
#define NB_TOT 7680
// Padded bias staging offsets (floats)
#define BI_K1 0
#define BI_K2 16
#define BI_S1 48
#define BI_S2 112
#define BI_LO 144
#define BI_UP 176
#define NBI_TOT 208

#if defined(__gfx1250__) && __has_builtin(__builtin_amdgcn_global_load_async_to_lds_b32)
#define HAVE_ASYNC_LDS 1
#else
#define HAVE_ASYNC_LDS 0
#endif

__device__ __forceinline__ v16h cat16(v8h a, v8h b) {
  return __builtin_shufflevector(a, b, 0,1,2,3,4,5,6,7,8,9,10,11,12,13,14,15);
}
// Load one wave-operand (16 halves) from a padded row: halves [8*hi,8*hi+8) and [16+8*hi,+8)
__device__ __forceinline__ v16h ldop(const _Float16* rowBase, int hi) {
  v8h g0 = *(const v8h*)(rowBase + 8*hi);
  v8h g1 = *(const v8h*)(rowBase + 16 + 8*hi);
  return cat16(g0, g1);
}
__device__ __forceinline__ v8f wmma16(v16h a, v16h b, v8f c) {
  return __builtin_amdgcn_wmma_f32_16x16x32_f16(false, a, false, b, (short)0, c, false, false);
}
__device__ __forceinline__ float sigm(float x) { return 1.0f / (1.0f + __expf(-x)); }

__global__ __launch_bounds__(128)
void cika_fused(const float* __restrict__ lower, const float* __restrict__ upper,
                const float* __restrict__ kca_dw_w, const float* __restrict__ kca_dw_b,
                const float* __restrict__ kca_m1_w, const float* __restrict__ kca_m1_b,
                const float* __restrict__ kca_m2_w, const float* __restrict__ kca_m2_b,
                const float* __restrict__ ksa_dw_w, const float* __restrict__ ksa_dw_b,
                const float* __restrict__ ksa_m1_w, const float* __restrict__ ksa_m1_b,
                const float* __restrict__ ksa_m2_w, const float* __restrict__ ksa_m2_b,
                const float* __restrict__ dyn_w,    const float* __restrict__ dyn_b,
                const float* __restrict__ low_pw_w, const float* __restrict__ low_pw_b,
                const float* __restrict__ up_dw_w,  const float* __restrict__ up_dw_b,
                const float* __restrict__ up_pw_w,  const float* __restrict__ up_pw_b,
                float* __restrict__ out_low, float* __restrict__ out_up)
{
  // LDS: 10240*2 + 13312 + 8192 + 4096 + 15360 + 832 = 62272 bytes
  __shared__ __align__(16) _Float16 sLow[TELEM];
  __shared__ __align__(16) _Float16 sUp [TELEM];
  __shared__ __align__(16) float    sW  [3328];          // dw/dyn weights + biases (f32)
  __shared__ __align__(16) _Float16 sMid[4][16][64];     // per-wave padded GEMM staging
  __shared__ __align__(16) _Float16 sKca[4][16][32];     // per-wave kca gate
  __shared__ __align__(16) _Float16 sB  [NB_TOT];        // padded, pre-cvt f16 B matrices
  __shared__ __align__(16) float    sBias[NBI_TOT];      // padded biases

  const int tid  = threadIdx.x;
  const int wave = tid >> 5;            // wave32
  const int lane = tid & 31;
  const int px   = lane & 15;           // pixel (M) this lane owns
  const int hi   = lane >> 4;
  const int nb   = blockIdx.z;
  const int y0   = blockIdx.y * TH;
  const int x0   = blockIdx.x * TW;

  // ---- 1) kick off async DMA of the f32 dw/dyn weight block into LDS ----
  for (int e = tid; e < 3328; e += 128) {
    const float* src; int off;
    if      (e <  800) { src = kca_dw_w; off = e;        }
    else if (e <  832) { src = kca_dw_b; off = e -  800; }
    else if (e < 1632) { src = ksa_dw_w; off = e -  832; }
    else if (e < 1664) { src = ksa_dw_b; off = e - 1632; }
    else if (e < 2464) { src = up_dw_w;  off = e - 1664; }
    else if (e < 2496) { src = up_dw_b;  off = e - 2464; }
    else if (e < 3296) { src = dyn_w;    off = e - 2496; }
    else               { src = dyn_b;    off = e - 3296; }
#if HAVE_ASYNC_LDS
    __builtin_amdgcn_global_load_async_to_lds_b32(
        (__attribute__((address_space(1))) int*)(src + off),
        (__attribute__((address_space(3))) int*)(&sW[e]), 0, 0);
#else
    sW[e] = src[off];
#endif
  }

  // ---- 2) halo tiles of lower/upper -> LDS f16 (overlaps with the async DMA) ----
  const size_t plane = (size_t)nb * CHN * IMG * IMG;
  const float* lo0 = lower + plane;
  const float* up0 = upper + plane;
  const bool interior = (y0 >= HALO) && (y0 + TH + HALO <= IMG) &&
                        (x0 >= HALO) && (x0 + TW + HALO <= IMG);
  if (interior) {
    for (int e = tid; e < TELEM; e += 128) {
      int c  = e / (TROWS*TCOLS);
      int r  = e % (TROWS*TCOLS);
      int ly = r / TCOLS, lx = r % TCOLS;
      size_t g = ((size_t)c*IMG + (y0 - HALO + ly))*IMG + (x0 - HALO + lx);
      sLow[e] = (_Float16)lo0[g];
      sUp [e] = (_Float16)up0[g];
    }
  } else {
    for (int e = tid; e < TELEM; e += 128) {
      int c  = e / (TROWS*TCOLS);
      int r  = e % (TROWS*TCOLS);
      int ly = r / TCOLS, lx = r % TCOLS;
      int gy = y0 - HALO + ly, gx = x0 - HALO + lx;
      float lo = 0.f, up = 0.f;
      if ((unsigned)gy < IMG && (unsigned)gx < IMG) {
        size_t g = ((size_t)c*IMG + gy)*IMG + gx;
        lo = lo0[g]; up = up0[g];
      }
      sLow[e] = (_Float16)lo; sUp[e] = (_Float16)up;
    }
  }

  // ---- 3) padded f16 B matrices + padded f32 biases -> LDS ----
  for (int e = tid; e < NB_TOT; e += 128) {
    float v = 0.f;
    if      (e < OB_K2) { int i=e-OB_K1, n=i>>5, k=i&31; if (n <  8) v = kca_m1_w[n*32+k]; }
    else if (e < OB_S1) { int i=e-OB_K2, n=i>>5, k=i&31; if (k <  8) v = kca_m2_w[n* 8+k]; }
    else if (e < OB_S2) { int i=e-OB_S1, n=i>>5, k=i&31; if (n < 50) v = ksa_m1_w[n*32+k]; }
    else if (e < OB_LO) { int i=e-OB_S2, n=i>>6, k=i&63; if (n < 25 && k < 50) v = ksa_m2_w[n*50+k]; }
    else if (e < OB_UP) { v = low_pw_w[e-OB_LO]; }
    else                { v = up_pw_w [e-OB_UP]; }
    sB[e] = (_Float16)v;
  }
  for (int e = tid; e < NBI_TOT; e += 128) {
    float v = 0.f;
    if      (e < BI_K2) { int i=e-BI_K1; if (i <  8) v = kca_m1_b[i]; }
    else if (e < BI_S1) { v = kca_m2_b[e-BI_K2]; }
    else if (e < BI_S2) { int i=e-BI_S1; if (i < 50) v = ksa_m1_b[i]; }
    else if (e < BI_LO) { int i=e-BI_S2; if (i < 25) v = ksa_m2_b[i]; }
    else if (e < BI_UP) { v = low_pw_b[e-BI_LO]; }
    else                { v = up_pw_b [e-BI_UP]; }
    sBias[e] = v;
  }

#if HAVE_ASYNC_LDS
#if __has_builtin(__builtin_amdgcn_s_wait_asynccnt)
  __builtin_amdgcn_s_wait_asynccnt(0);
#else
  asm volatile("s_wait_asynccnt 0" ::: "memory");
#endif
#endif
  __syncthreads();

  const int row = wave;

  // ---- pre-zero this wave's padded staging (keeps K-padding zero everywhere) ----
  {
    _Float16* mb = &sMid[wave][0][0];
    v8h z = {};
    #pragma unroll
    for (int e = 0; e < 4; ++e)
      *(v8h*)(mb + (lane + 32*e) * 8) = z;   // 16*64 halves
  }

  // ---- depthwise 5x5 taps: results drop straight into WMMA-A layout ----
  v16h aK, aS; float updw[16];
  #pragma unroll
  for (int t = 0; t < 16; ++t) {
    int c = (t < 8) ? (hi*8 + t) : (16 + hi*8 + (t - 8));   // lane's K/channel slot
    const _Float16* tl = &sLow[(c*TROWS + row)*TCOLS + px];
    const _Float16* tu = &sUp [(c*TROWS + row)*TCOLS + px];
    const float* wk = &sW[c*25];
    const float* ws = &sW[832  + c*25];
    const float* wu = &sW[1664 + c*25];
    float sk = sW[800 + c], ss = sW[1632 + c], su = sW[2464 + c];
    for (int i = 0; i < 5; ++i)
      for (int j = 0; j < 5; ++j) {
        float xl = (float)tl[i*TCOLS + j];
        float xu = (float)tu[i*TCOLS + j];
        sk = fmaf(xl, wk[i*5+j], sk);
        ss = fmaf(xu, ws[i*5+j], ss);
        su = fmaf(xu, wu[i*5+j], su);
      }
    aK[t]   = (_Float16)fmaxf(sk, 0.f);
    aS[t]   = (_Float16)fmaxf(ss, 0.f);
    updw[t] = su;
  }

  // ================= KCA chain (all unconditional) =================
  { // m1: 32 -> 8 (padded N=16)
    v8f acc = {};
    acc = wmma16(aK, ldop(&sB[OB_K1 + px*32], hi), acc);
    float bias = sBias[BI_K1 + px];
    #pragma unroll
    for (int r = 0; r < 8; ++r)
      sMid[wave][r + 8*hi][px] = (_Float16)fmaxf(acc[r] + bias, 0.f);
  }
  { // m2: 8 (K padded to 32) -> 32, sigmoid -> sKca
    v16h a2 = ldop(&sMid[wave][px][0], hi);
    #pragma unroll
    for (int nt = 0; nt < 2; ++nt) {
      int ng = nt*16 + px;
      v8f acc = {};
      acc = wmma16(a2, ldop(&sB[OB_K2 + ng*32], hi), acc);
      float bias = sBias[BI_K2 + ng];
      #pragma unroll
      for (int r = 0; r < 8; ++r)
        sKca[wave][r + 8*hi][ng] = (_Float16)sigm(acc[r] + bias);
    }
  }

  // ================= KSA chain =================
  #pragma unroll
  for (int nt = 0; nt < 4; ++nt) {   // m1: 32 -> 50 (padded N=64)
    int ng = nt*16 + px;
    v8f acc = {};
    acc = wmma16(aS, ldop(&sB[OB_S1 + ng*32], hi), acc);
    float bias = sBias[BI_S1 + ng];            // zero in padding -> keeps staging zero
    #pragma unroll
    for (int r = 0; r < 8; ++r)
      sMid[wave][r + 8*hi][ng] = (_Float16)fmaxf(acc[r] + bias, 0.f);
  }
  { // m2: 50 (two K=32 chunks) -> 25 (padded N=32)
    v16h aq0 = ldop(&sMid[wave][px][0],  hi);
    v16h aq1 = ldop(&sMid[wave][px][32], hi);
    #pragma unroll
    for (int nt = 0; nt < 2; ++nt) {
      int ng = nt*16 + px;
      v8f acc = {};
      acc = wmma16(aq0, ldop(&sB[OB_S2 + ng*64],      hi), acc);
      acc = wmma16(aq1, ldop(&sB[OB_S2 + ng*64 + 32], hi), acc);
      float bias = sBias[BI_S2 + ng];
      #pragma unroll
      for (int r = 0; r < 8; ++r)
        sMid[wave][r + 8*hi][ng] = (_Float16)sigm(acc[r] + bias);  // reuse as ksa[0..24]
    }
  }

  // ================= low path: dynamic depthwise + 1x1 =================
  float rk[25];
  {
    const _Float16* m = &sMid[wave][px][0];
    v8h r0 = *(const v8h*)(m);
    v8h r1 = *(const v8h*)(m + 8);
    v8h r2 = *(const v8h*)(m + 16);
    #pragma unroll
    for (int i = 0; i < 8; ++i) { rk[i] = (float)r0[i]; rk[8+i] = (float)r1[i]; }
    #pragma unroll
    for (int i = 0; i < 8; ++i) if (16+i < 25) rk[16+i] = (float)r2[i];
    rk[24] = (float)m[24];
  }
  v16h aD;
  #pragma unroll
  for (int t = 0; t < 16; ++t) {
    int c = (t < 8) ? (hi*8 + t) : (16 + hi*8 + (t - 8));
    const _Float16* tl = &sLow[(c*TROWS + row)*TCOLS + px];
    const float* wd = &sW[2496 + c*25];
    float s = sW[3296 + c];
    for (int i = 0; i < 5; ++i)
      for (int j = 0; j < 5; ++j)
        s = fmaf((float)tl[i*TCOLS + j] * rk[i*5+j], wd[i*5+j], s);
    aD[t] = (_Float16)s;
  }
  {
    const int y = y0 + row;
    #pragma unroll
    for (int nt = 0; nt < 2; ++nt) {
      int ng = nt*16 + px;
      v8f acc = {};
      acc = wmma16(aD, ldop(&sB[OB_LO + ng*32], hi), acc);
      float bias = sBias[BI_LO + ng];
      float* dst = out_low + ((size_t)(nb*CHN + ng)*IMG + y)*IMG + x0 + 8*hi;
      v4f o0 = {acc[0]+bias, acc[1]+bias, acc[2]+bias, acc[3]+bias};
      v4f o1 = {acc[4]+bias, acc[5]+bias, acc[6]+bias, acc[7]+bias};
      *(v4f*)(dst)     = o0;
      *(v4f*)(dst + 4) = o1;
    }
  }

  // ================= up path: gate dw by kca + 1x1 =================
  v16h aU;
  {
    v8h g0 = *(const v8h*)(&sKca[wave][px][8*hi]);
    v8h g1 = *(const v8h*)(&sKca[wave][px][16 + 8*hi]);
    #pragma unroll
    for (int t = 0; t < 8; ++t) {
      aU[t]     = (_Float16)(updw[t]     * (float)g0[t]);
      aU[8 + t] = (_Float16)(updw[8 + t] * (float)g1[t]);
    }
  }
  {
    const int y = y0 + row;
    #pragma unroll
    for (int nt = 0; nt < 2; ++nt) {
      int ng = nt*16 + px;
      v8f acc = {};
      acc = wmma16(aU, ldop(&sB[OB_UP + ng*32], hi), acc);
      float bias = sBias[BI_UP + ng];
      float* dst = out_up + ((size_t)(nb*CHN + ng)*IMG + y)*IMG + x0 + 8*hi;
      v4f o0 = {acc[0]+bias, acc[1]+bias, acc[2]+bias, acc[3]+bias};
      v4f o1 = {acc[4]+bias, acc[5]+bias, acc[6]+bias, acc[7]+bias};
      *(v4f*)(dst)     = o0;
      *(v4f*)(dst + 4) = o1;
    }
  }
}

extern "C" void kernel_launch(void* const* d_in, const int* in_sizes, int n_in,
                              void* d_out, int out_size, void* d_ws, size_t ws_size,
                              hipStream_t stream) {
  (void)n_in; (void)d_ws; (void)ws_size; (void)out_size;
  const float* lower    = (const float*)d_in[0];
  const float* upper    = (const float*)d_in[1];
  const float* kca_dw_w = (const float*)d_in[2];
  const float* kca_dw_b = (const float*)d_in[3];
  const float* kca_m1_w = (const float*)d_in[4];
  const float* kca_m1_b = (const float*)d_in[5];
  const float* kca_m2_w = (const float*)d_in[6];
  const float* kca_m2_b = (const float*)d_in[7];
  const float* ksa_dw_w = (const float*)d_in[8];
  const float* ksa_dw_b = (const float*)d_in[9];
  const float* ksa_m1_w = (const float*)d_in[10];
  const float* ksa_m1_b = (const float*)d_in[11];
  const float* ksa_m2_w = (const float*)d_in[12];
  const float* ksa_m2_b = (const float*)d_in[13];
  const float* dyn_w    = (const float*)d_in[14];
  const float* dyn_b    = (const float*)d_in[15];
  const float* low_pw_w = (const float*)d_in[16];
  const float* low_pw_b = (const float*)d_in[17];
  const float* up_dw_w  = (const float*)d_in[18];
  const float* up_dw_b  = (const float*)d_in[19];
  const float* up_pw_w  = (const float*)d_in[20];
  const float* up_pw_b  = (const float*)d_in[21];

  const int N = in_sizes[0] / (CHN * IMG * IMG);   // 4
  float* out_low = (float*)d_out;
  float* out_up  = out_low + (size_t)N * CHN * IMG * IMG;

  dim3 grid(IMG / TW, IMG / TH, N);   // 16 x 64 x N
  cika_fused<<<grid, 128, 0, stream>>>(
      lower, upper,
      kca_dw_w, kca_dw_b, kca_m1_w, kca_m1_b, kca_m2_w, kca_m2_b,
      ksa_dw_w, ksa_dw_b, ksa_m1_w, ksa_m1_b, ksa_m2_w, ksa_m2_b,
      dyn_w, dyn_b, low_pw_w, low_pw_b,
      up_dw_w, up_dw_b, up_pw_w, up_pw_b,
      out_low, out_up);
}
